// ESN_4458176053698
// MI455X (gfx1250) — matrix operand order repaired and checked by reading, not compiled, not run on hardware
//
#include <hip/hip_runtime.h>
#include <math.h>

// ---------------------------------------------------------------------------
// ESN on MI455X (gfx1250, wave32, WMMA).
//   B=16, T=1000, D=256, N=2000.
// Phase 1 (prep): x -> bf16 [T][16][256]; Wout -> bf16 B-layout; h0 -> bf16.
// Phase 2 (persistent recurrence): 125 WGs (one 16-col tile each), Wres slice
//   (rho folded) + Win slice (omega folded) staged in LDS once; per step each
//   wave does 16 v_wmma_f32_16x16x32_bf16 over its K slab, partials reduced in
//   LDS, wave 0 applies tanh + leaky update, ping-pong h in global, grid-wide
//   atomic-counter barrier between steps.
// Phase 3: readout GEMM H[16000x2048]@WoutT via WMMA, 64 wmma per 16x16 tile.
// ---------------------------------------------------------------------------

#define B_   16
#define T_   1000
#define D_   256
#define N_   2000
#define KP   2048            // padded K / global h & H row stride (elements)
#define KPB  2056            // LDS / WoutT B-matrix row stride (bank-spread pad)
#define KPW  264             // LDS Win B-matrix row stride
#define NWG  125             // N/16 column tiles

typedef __attribute__((ext_vector_type(16))) __bf16      v16bf;
typedef __attribute__((ext_vector_type(8)))  float       v8f;
typedef __attribute__((ext_vector_type(4)))  unsigned int u32x4;
typedef __attribute__((ext_vector_type(4)))  float       f32x4;

union Frag { v16bf v; u32x4 q[2]; };

// ---------------------------------------------------------------- prep kernels
__global__ void esn_prep_x(const float* __restrict__ x, __bf16* __restrict__ xT) {
    // x: [B][T][D] f32  ->  xT: [T][B][D] bf16 (A-matrix friendly, contiguous per t)
    long long i = (long long)blockIdx.x * blockDim.x + threadIdx.x;
    if (i >= (long long)B_ * T_ * D_) return;
    int d = (int)(i % D_);
    long long r = i / D_;
    int t = (int)(r % T_);
    int b = (int)(r / T_);
    xT[((long long)t * B_ + b) * D_ + d] = (__bf16)x[i];
}

__global__ void esn_prep_wout(const float* __restrict__ Wout, __bf16* __restrict__ WoutT) {
    // Wout: [N][D] f32 -> WoutT: [D][KPB] bf16, WoutT[n][k] = Wout[k][n], zero pad
    long long i = (long long)blockIdx.x * blockDim.x + threadIdx.x;
    if (i >= (long long)D_ * KPB) return;
    int k = (int)(i % KPB);
    int n = (int)(i / KPB);
    float v = (k < N_) ? Wout[(long long)k * D_ + n] : 0.0f;
    WoutT[i] = (__bf16)v;
}

__global__ void esn_prep_h(const float* __restrict__ h0, __bf16* __restrict__ hbuf) {
    // hbuf: [2][B][KP] bf16; buf0 <- h0 (pad 0), buf1 <- 0 (pad cols stay 0 forever)
    int i = blockIdx.x * blockDim.x + threadIdx.x;
    if (i >= 2 * B_ * KP) return;
    int k = i % KP;
    int r = (i / KP) % B_;
    int buf = i / (B_ * KP);
    float v = (buf == 0 && k < N_) ? h0[r * N_ + k] : 0.0f;
    hbuf[i] = (__bf16)v;
}

// ----------------------------------------------------- persistent recurrence
__global__ __launch_bounds__(128, 1)
void esn_recurrence(const float* __restrict__ Wres,   // [N][N] f32
                    const float* __restrict__ Win,    // [D][N] f32
                    const float* __restrict__ omega_p,
                    const float* __restrict__ rho_p,
                    const float* __restrict__ alpha_p,
                    const __bf16* __restrict__ xT,    // [T][B][D] bf16
                    __bf16* __restrict__ hbuf,        // [2][B][KP] bf16 ping-pong
                    __bf16* __restrict__ H,           // [T][B][KP] bf16 history
                    unsigned int* __restrict__ counter)
{
    extern __shared__ char smem[];
    __bf16* Bt   = (__bf16*)smem;                                  // [16][KPB] rho*Wres slice
    __bf16* Wt   = (__bf16*)(smem + 16 * KPB * 2);                 // [16][KPW] omega*Win slice
    float*  part = (float*)(smem + 16 * KPB * 2 + 16 * KPW * 2);   // [4][32][8] partial C

    const int tid   = threadIdx.x;
    const int lane  = tid & 31;
    const int wave  = tid >> 5;
    const int jbase = blockIdx.x * 16;

    const float omega = omega_p[0];
    const float rho   = rho_p[0];
    const float alpha = alpha_p[0];

    // ---- one-time: stage B matrices (transposed, scaled, bf16) into LDS ----
    {
        const int n = tid & 15;
        for (int k = tid >> 4; k < KPB; k += 8) {
            float v = (k < N_) ? rho * Wres[(long long)k * N_ + jbase + n] : 0.0f;
            Bt[n * KPB + k] = (__bf16)v;
        }
        for (int k = tid >> 4; k < KPW; k += 8) {
            float v = (k < D_) ? omega * Win[(long long)k * N_ + jbase + n] : 0.0f;
            Wt[n * KPW + k] = (__bf16)v;
        }
    }
    __syncthreads();

    const int m  = lane & 15;         // A row (batch) == B column within tile
    const int kh = (lane >> 4) * 8;   // unused; A fragments also use kb
    const int kb = (lane >> 4) * 16;  // B-fragment K subgroup offset

    for (int t = 0; t < T_; ++t) {
        const __bf16* cur = hbuf + (long long)(t & 1) * B_ * KP;
        __bf16*       nxt = hbuf + (long long)((t + 1) & 1) * B_ * KP;

        v8f acc = {};

        // ---- reservoir term: this wave's K slab of rho*(h @ Wres) ----
        const int kslab = wave * (KP / 4);   // 512 per wave
        #pragma unroll 4
        for (int kk = 0; kk < KP / 4; kk += 32) {
            const int k0 = kslab + kk;
            Frag a, b;
            const __bf16* ap = cur + m * KP + k0 + kb;
            a.q[0] = *(const u32x4*)(ap);
            a.q[1] = *(const u32x4*)(ap + 8);
            const __bf16* bp = Bt + m * KPB + k0 + kb;
            b.q[0] = *(const u32x4*)(bp);
            b.q[1] = *(const u32x4*)(bp + 8);
            acc = __builtin_amdgcn_wmma_f32_16x16x32_bf16(
                      false, a.v, false, b.v, (short)0, acc, false, false);
        }

        if (wave != 0) {
            float* p = part + (wave * 32 + lane) * 8;
            f32x4 lo, hi;
            lo[0] = acc[0]; lo[1] = acc[1]; lo[2] = acc[2]; lo[3] = acc[3];
            hi[0] = acc[4]; hi[1] = acc[5]; hi[2] = acc[6]; hi[3] = acc[7];
            *(f32x4*)(p)     = lo;
            *(f32x4*)(p + 4) = hi;
        } else {
            // ---- input term: omega * x_t @ Win (K = 256, 8 WMMAs) ----
            const __bf16* xrow = xT + ((long long)t * B_ + m) * D_;
            #pragma unroll
            for (int k0 = 0; k0 < D_; k0 += 32) {
                Frag a, b;
                a.q[0] = *(const u32x4*)(xrow + k0 + kb);
                a.q[1] = *(const u32x4*)(xrow + k0 + kb + 8);
                const __bf16* bp = Wt + m * KPW + k0 + kb;
                b.q[0] = *(const u32x4*)(bp);
                b.q[1] = *(const u32x4*)(bp + 8);
                acc = __builtin_amdgcn_wmma_f32_16x16x32_bf16(
                          false, a.v, false, b.v, (short)0, acc, false, false);
            }
        }
        __syncthreads();

        if (wave == 0) {
            // reduce K-slab partials from waves 1..3
            #pragma unroll
            for (int w = 1; w < 4; ++w) {
                const float* p = part + (w * 32 + lane) * 8;
                f32x4 lo = *(const f32x4*)(p);
                f32x4 hi = *(const f32x4*)(p + 4);
                acc[0] += lo[0]; acc[1] += lo[1]; acc[2] += lo[2]; acc[3] += lo[3];
                acc[4] += hi[0]; acc[5] += hi[1]; acc[6] += hi[2]; acc[7] += hi[3];
            }
            // activation + leaky residual update + publish h_{t+1}, H[t]
            const int col = jbase + (lane & 15);
            #pragma unroll
            for (int i = 0; i < 8; ++i) {
                const int row = (lane >> 4) * 8 + i;      // C/D layout: VGPR i
                float hold = (float)cur[row * KP + col];
                float cand = tanhf(acc[i]);
                float hnew = hold + alpha * (cand - hold);
                __bf16 hb = (__bf16)hnew;
                nxt[row * KP + col] = hb;
                H[((long long)t * B_ + row) * KP + col] = hb;
            }
        }

        // ---- grid-wide barrier (monotonic counter; 125 WGs co-resident) ----
        __threadfence();
        __syncthreads();
        if (tid == 0) {
            __hip_atomic_fetch_add(counter, 1u, __ATOMIC_ACQ_REL, __HIP_MEMORY_SCOPE_AGENT);
            const unsigned target = (unsigned)(t + 1) * NWG;
            while (__hip_atomic_load(counter, __ATOMIC_ACQUIRE, __HIP_MEMORY_SCOPE_AGENT) < target)
                __builtin_amdgcn_s_sleep(1);
        }
        __syncthreads();
    }
}

// ------------------------------------------------------------------- readout
__global__ __launch_bounds__(32, 8)
void esn_readout(const __bf16* __restrict__ H,      // [T][B][KP] bf16
                 const __bf16* __restrict__ WoutT,  // [D][KPB] bf16
                 float* __restrict__ out)           // [B][T][D] f32
{
    const int lane = threadIdx.x;
    const int t  = blockIdx.y;       // M tile == timestep (16 batch rows)
    const int nt = blockIdx.x;       // 16 output cols per tile, D/16 = 16 tiles
    const int m  = lane & 15;
    const int kh = (lane >> 4) * 8;
    const int kb = (lane >> 4) * 16;

    const __bf16* arow = H + ((long long)t * B_ + m) * KP;
    const __bf16* brow = WoutT + (long long)(nt * 16 + m) * KPB;

    v8f acc = {};
    #pragma unroll 8
    for (int k0 = 0; k0 < KP; k0 += 32) {
        Frag a, b;
        a.q[0] = *(const u32x4*)(arow + k0 + kb);
        a.q[1] = *(const u32x4*)(arow + k0 + kb + 8);
        b.q[0] = *(const u32x4*)(brow + k0 + kb);
        b.q[1] = *(const u32x4*)(brow + k0 + kb + 8);
        acc = __builtin_amdgcn_wmma_f32_16x16x32_bf16(
                  false, a.v, false, b.v, (short)0, acc, false, false);
    }

    const int d = nt * 16 + (lane & 15);
    #pragma unroll
    for (int i = 0; i < 8; ++i) {
        const int b = (lane >> 4) * 8 + i;   // tile row == batch index
        out[((long long)b * T_ + t) * D_ + d] = acc[i];
    }
}

// -------------------------------------------------------------------- launch
extern "C" void kernel_launch(void* const* d_in, const int* in_sizes, int n_in,
                              void* d_out, int out_size, void* d_ws, size_t ws_size,
                              hipStream_t stream) {
    (void)in_sizes; (void)n_in; (void)out_size; (void)ws_size;

    const float* x     = (const float*)d_in[0];   // [B,T,D]
    const float* Win   = (const float*)d_in[1];   // [D,N]
    const float* Wres  = (const float*)d_in[2];   // [N,N]
    const float* Wout  = (const float*)d_in[3];   // [N,D]
    const float* h0    = (const float*)d_in[4];   // [B,N]
    const float* omega = (const float*)d_in[5];
    const float* rho   = (const float*)d_in[6];
    const float* alpha = (const float*)d_in[7];
    float* out = (float*)d_out;

    // workspace layout (256-byte aligned sections)
    char* ws = (char*)d_ws;
    unsigned int* counter = (unsigned int*)ws;                       size_t off = 256;
    __bf16* xT    = (__bf16*)(ws + off);  off += (size_t)T_ * B_ * D_ * 2;   // 8,192,000
    __bf16* hbuf  = (__bf16*)(ws + off);  off += (size_t)2 * B_ * KP * 2;    //   131,072
    __bf16* WoutT = (__bf16*)(ws + off);  off += (size_t)D_ * KPB * 2;       // 1,052,672
    __bf16* Hbuf  = (__bf16*)(ws + off);                                     // 65,536,000

    hipMemsetAsync(counter, 0, 256, stream);
    hipMemsetAsync(Hbuf, 0, (size_t)T_ * B_ * KP * 2, stream);   // zero K-pad columns

    {
        long long n = (long long)B_ * T_ * D_;
        esn_prep_x<<<(unsigned)((n + 255) / 256), 256, 0, stream>>>(x, xT);
    }
    {
        long long n = (long long)D_ * KPB;
        esn_prep_wout<<<(unsigned)((n + 255) / 256), 256, 0, stream>>>(Wout, WoutT);
    }
    {
        int n = 2 * B_ * KP;
        esn_prep_h<<<(n + 255) / 256, 256, 0, stream>>>(h0, hbuf);
    }

    const size_t smem = (size_t)16 * KPB * 2 + (size_t)16 * KPW * 2 + 4 * 32 * 8 * 4; // 78,336 B
    esn_recurrence<<<NWG, 128, smem, stream>>>(Wres, Win, omega, rho, alpha,
                                               xT, hbuf, Hbuf, counter);

    esn_readout<<<dim3(D_ / 16, T_), 32, 0, stream>>>(Hbuf, WoutT, out);
}
